// Arch11GraphEncoder_67886253081063
// MI455X (gfx1250) — matrix operand order repaired and checked by reading, not compiled, not run on hardware
//
#include <hip/hip_runtime.h>

// ---------------- problem constants ----------------
#define HD      256          // hidden size H
#define LLAYERS 6
#define NHEADS  4
#define DHEAD   64
#define NTOT    4096
#define MVIEWS  4
#define KSUB    8
#define SSUB    16384        // NTOT * MVIEWS
#define FN      131072       // SSUB * KSUB
#define EINTRA  262144
#define EGLOB   65536
#define NATOM   64
#define NBOND   8
#define NGRAPH  64
#define EPSV    1e-5f

typedef __attribute__((ext_vector_type(16))) __bf16 v16bf;
typedef __attribute__((ext_vector_type(8)))  __bf16 v8bf;
typedef __attribute__((ext_vector_type(8)))  float  v8f;
typedef __attribute__((ext_vector_type(4)))  unsigned int u32x4;
typedef __attribute__((ext_vector_type(4)))  int i32x4;
typedef __attribute__((ext_vector_type(8)))  int i32x8;

#if defined(__HIP_DEVICE_COMPILE__) && __has_builtin(__builtin_amdgcn_tensor_load_to_lds)
#define HAVE_TDM 1
#else
#define HAVE_TDM 0
#endif

__device__ __forceinline__ __bf16 f2bf(float f) {
  unsigned u = __float_as_uint(f);
  unsigned r = (u + 0x7FFFu + ((u >> 16) & 1u)) >> 16;
  unsigned short s = (unsigned short)r;
  return __builtin_bit_cast(__bf16, s);
}

__device__ __forceinline__ float geluf(float x) {
  return 0.5f * x * (1.0f + erff(x * 0.70710678118654752f));
}

// ---------------------------------------------------------------------------
// WMMA bf16 GEMM:  C[M,N] = act(A[M,K] @ W[K,N] + bias[N] (+ resid[M,N]))
// block = 256 threads = 8 waves; block tile 128(M) x 64(N); K step 32.
// wave w: wm = w%4 -> rows 32*wm..+32 ; wn = w/4 -> cols 32*wn..+32
// each wave: FOUR v_wmma_f32_16x16x32_bf16 per K step (2 A frags x 2 B frags).
// act: 0=none 1=relu 2=gelu
// ---------------------------------------------------------------------------
__global__ __launch_bounds__(256)
void k_gemm_wmma(const __bf16* __restrict__ A, const __bf16* __restrict__ W,
                 const float* __restrict__ bias, const float* __restrict__ resid,
                 float* __restrict__ Cf, __bf16* __restrict__ Cb,
                 int Mm, int Nn, int Kk, int act) {
  __shared__ __align__(32) __bf16 lA[128 * 32];
  __shared__ __align__(32) __bf16 lB[32 * 64];

  const int tid   = threadIdx.x;
  const int lane  = tid & 31;
  const int w     = tid >> 5;
  const int wm    = w & 3;       // 0..3
  const int wn    = w >> 2;      // 0..1
  const int m0    = blockIdx.y * 128;
  const int n0    = blockIdx.x * 64;

  v8f acc00 = {}, acc01 = {}, acc10 = {}, acc11 = {};

  const int arow = tid >> 1;            // 0..127
  const int aseg = (tid & 1) * 16;      // 0 / 16
  const int brow = tid >> 3;            // 0..31
  const int bseg = (tid & 7) * 8;       // 0..56

  for (int k0 = 0; k0 < Kk; k0 += 32) {
    // stage A tile (128x32) : 32B per thread
    {
      const __bf16* src = A + (size_t)(m0 + arow) * Kk + k0 + aseg;
      __bf16* dst = &lA[arow * 32 + aseg];
      *(uint4*)(dst)     = *(const uint4*)(src);
      *(uint4*)(dst + 8) = *(const uint4*)(src + 8);
      __builtin_prefetch(src + 64, 0, 0);                    // global_prefetch_b8
    }
    // stage B tile (32x64) : 16B per thread
    {
      const __bf16* src = W + (size_t)(k0 + brow) * Nn + n0 + bseg;
      *(uint4*)(&lB[brow * 64 + bseg]) = *(const uint4*)(src);
      __builtin_prefetch(src + (size_t)64 * Nn, 0, 0);
    }
    __syncthreads();

    // B fragments: 32x16 bf16 -> lane l holds col (l&15), K in same order as A
    v16bf bf0, bf1;
#pragma unroll
    for (int s = 0; s < 16; ++s) {
      const int kb = (s < 8) ? ((lane >> 4) * 8 + s) : (16 + (lane >> 4) * 8 + (s - 8));
      bf0[s] = lB[kb * 64 + wn * 32 + (lane & 15)];
      bf1[s] = lB[kb * 64 + wn * 32 + 16 + (lane & 15)];
    }

    // A fragments: 16x32 bf16 per ISA layout:
    // lane l -> row (l&15); holds K = h2..h2+7 and 16+h2..16+h2+7, h2 = (l>>4)*8
    const int h2 = (lane >> 4) * 8;
    union { v16bf v; v8bf h[2]; } a0, a1;
    {
      const __bf16* r0 = &lA[(wm * 32 + (lane & 15)) * 32];
      a0.h[0] = *(const v8bf*)(r0 + h2);
      a0.h[1] = *(const v8bf*)(r0 + 16 + h2);
      const __bf16* r1 = &lA[(wm * 32 + 16 + (lane & 15)) * 32];
      a1.h[0] = *(const v8bf*)(r1 + h2);
      a1.h[1] = *(const v8bf*)(r1 + 16 + h2);
    }

    acc00 = __builtin_amdgcn_wmma_f32_16x16x32_bf16(false, a0.v, false, bf0,
                                                    (short)0, acc00, false, false);
    acc01 = __builtin_amdgcn_wmma_f32_16x16x32_bf16(false, a0.v, false, bf1,
                                                    (short)0, acc01, false, false);
    acc10 = __builtin_amdgcn_wmma_f32_16x16x32_bf16(false, a1.v, false, bf0,
                                                    (short)0, acc10, false, false);
    acc11 = __builtin_amdgcn_wmma_f32_16x16x32_bf16(false, a1.v, false, bf1,
                                                    (short)0, acc11, false, false);
    __syncthreads();
  }

  // C layout: lane l, VGPR r -> row = half*8 + r (half = l>>4), col = l&15
  const int half = lane >> 4;
#pragma unroll
  for (int tn = 0; tn < 2; ++tn) {
    const int col = n0 + wn * 32 + tn * 16 + (lane & 15);
    const float bv = bias ? bias[col] : 0.0f;
#pragma unroll
    for (int tm = 0; tm < 2; ++tm) {
      const v8f& acc = tm == 0 ? (tn == 0 ? acc00 : acc01) : (tn == 0 ? acc10 : acc11);
#pragma unroll
      for (int r = 0; r < 8; ++r) {
        int row = m0 + wm * 32 + tm * 16 + half * 8 + r;
        float v = acc[r] + bv;
        size_t idx = (size_t)row * Nn + col;
        if (resid) v += resid[idx];
        if (act == 1) v = fmaxf(v, 0.0f);
        else if (act == 2) v = geluf(v);
        if (Cf) Cf[idx] = v;
        if (Cb) Cb[idx] = f2bf(v);
      }
    }
  }
}

// ---------------------------------------------------------------------------
// TDM tile copy: each block copies 32 rows x 256 f32 through LDS using the
// Tensor Data Mover (tensor_load_to_lds + tensor_store_from_lds, TENSORcnt).
// D# per CDNA5 ISA: group0 {count=1, lds_addr, global_addr,
// type=2}; group1 {data_size=4B, tensor_dim0=256, tensor_dim1=32,
// tile_dim0=256, tile_dim1=32, tensor_dim0_stride=256}.
// ---------------------------------------------------------------------------
__global__ __launch_bounds__(32)
void k_copy_tdm(const float* __restrict__ src, float* __restrict__ dst) {
  __shared__ __align__(256) float tile[32 * HD];
  const size_t r0 = (size_t)blockIdx.x * 32;
  const float* s = src + r0 * HD;
  float* d = dst + r0 * HD;
#if HAVE_TDM
  unsigned lds = (unsigned)(unsigned long long)(void*)tile;   // LDS aperture low bits
  unsigned long long ga = (unsigned long long)s;
  u32x4 g0l = { 1u, lds, (unsigned)ga,
                ((unsigned)((ga >> 32) & 0x01FFFFFFull)) | (2u << 30) };
  i32x8 g1 = { (int)(2u << 16),               // data_size = 4B
               (int)(((unsigned)HD) << 16),   // tensor_dim0 lo16 (bits 63:48)
               (int)(32u << 16),              // tensor_dim1 lo16 (bits 95:80)
               (int)(((unsigned)HD) << 16),   // tile_dim0   (bits 127:112)
               32,                            // tile_dim1   (bits 143:128)
               HD,                            // tensor_dim0_stride lo32
               0, 0 };
  i32x4 z4 = {};
#if __clang_major__ >= 23
  i32x8 z8 = {};
  __builtin_amdgcn_tensor_load_to_lds(g0l, g1, z4, z4, z8, 0);
#else
  __builtin_amdgcn_tensor_load_to_lds(g0l, g1, z4, z4, 0);
#endif
  __builtin_amdgcn_s_wait_tensorcnt(0);
  unsigned long long gd = (unsigned long long)d;
  u32x4 g0s = { 1u, lds, (unsigned)gd,
                ((unsigned)((gd >> 32) & 0x01FFFFFFull)) | (2u << 30) };
#if __has_builtin(__builtin_amdgcn_tensor_store_from_lds)
#if __clang_major__ >= 23
  __builtin_amdgcn_tensor_store_from_lds(g0s, g1, z4, z4, z8, 0);
#else
  __builtin_amdgcn_tensor_store_from_lds(g0s, g1, z4, z4, 0);
#endif
  __builtin_amdgcn_s_wait_tensorcnt(0);
#else
  for (int i = threadIdx.x; i < 32 * HD; i += 32) d[i] = tile[i];
  (void)g0s;
#endif
#else
  (void)tile;
  for (int i = threadIdx.x; i < 32 * HD; i += 32) d[i] = s[i];
#endif
}

// ---------------- elementwise / utility kernels ----------------
__global__ void k_zero(float* p, long n) {
  long i = (long)blockIdx.x * 256 + threadIdx.x;
  if (i < n) p[i] = 0.0f;
}
__global__ void k_cvt_bf16(const float* s, __bf16* d, long n) {
  long i = (long)blockIdx.x * 256 + threadIdx.x;
  if (i < n) d[i] = f2bf(s[i]);
}

// h = (atom_emb[x[cid]] + role_emb[is_root]) * valid ; also bf16 copy
__global__ void k_init_h(const float* __restrict__ atom_emb,
                         const float* __restrict__ role_emb,
                         const int* __restrict__ xid,
                         const int* __restrict__ node_ids,
                         const unsigned char* __restrict__ valid,
                         float* __restrict__ h, __bf16* __restrict__ hb) {
  long i = (long)blockIdx.x * 256 + threadIdx.x;
  if (i >= (long)FN * HD) return;
  int f = (int)(i >> 8), j = (int)(i & 255);
  int cid = node_ids[f]; if (cid < 0) cid = 0;
  int isr = ((f & (KSUB - 1)) == 0) ? 1 : 0;
  float vf = valid[f] ? 1.0f : 0.0f;
  float v = (atom_emb[(size_t)xid[cid] * HD + j] + role_emb[(size_t)isr * HD + j]) * vf;
  h[i] = v; hb[i] = f2bf(v);
}

// edge message: agg[dst] += relu(h[src] + bond_emb[ea])
__global__ void k_edge_msg(const float* __restrict__ h, const float* __restrict__ bond,
                           const int* __restrict__ src, const int* __restrict__ dst,
                           const int* __restrict__ ea, float* __restrict__ agg, long nE) {
  long i = (long)blockIdx.x * 256 + threadIdx.x;
  if (i >= nE * HD) return;
  int e = (int)(i >> 8), j = (int)(i & 255);
  float m = h[(size_t)src[e] * HD + j] + bond[(size_t)ea[e] * HD + j];
  m = fmaxf(m, 0.0f);
  atomicAdd(&agg[(size_t)dst[e] * HD + j], m);
}

__global__ void k_counts(const int* __restrict__ node_ids,
                         const unsigned char* __restrict__ valid, float* __restrict__ cnt) {
  int f = blockIdx.x * 256 + threadIdx.x;
  if (f >= FN) return;
  int cid = node_ids[f]; if (cid < 0) cid = 0;
  atomicAdd(&cnt[cid], valid[f] ? 1.0f : 0.0f);
}

__global__ void k_segsum(const float* __restrict__ h, const int* __restrict__ node_ids,
                         float* __restrict__ xsum) {
  long i = (long)blockIdx.x * 256 + threadIdx.x;
  if (i >= (long)FN * HD) return;
  int f = (int)(i >> 8), j = (int)(i & 255);
  int cid = node_ids[f]; if (cid < 0) cid = 0;
  atomicAdd(&xsum[(size_t)cid * HD + j], h[i]);
}

__global__ void k_divcnt(float* __restrict__ xsum, const float* __restrict__ cnt,
                         __bf16* __restrict__ outb) {
  long i = (long)blockIdx.x * 256 + threadIdx.x;
  if (i >= (long)NTOT * HD) return;
  int n = (int)(i >> 8);
  float v = xsum[i] / fmaxf(cnt[n], 1.0f);
  xsum[i] = v;
  if (outb) outb[i] = f2bf(v);
}

// batchnorm stats: one block per column
__global__ void k_bn_stats(const float* __restrict__ x, int Mrows,
                           float* __restrict__ mu, float* __restrict__ rstd) {
  int j = blockIdx.x, t = threadIdx.x;
  float s = 0.0f, q = 0.0f;
  for (int r = t; r < Mrows; r += 256) {
    float v = x[(size_t)r * HD + j];
    s += v; q += v * v;
  }
  __shared__ float s1[256], s2[256];
  s1[t] = s; s2[t] = q; __syncthreads();
  for (int off = 128; off > 0; off >>= 1) {
    if (t < off) { s1[t] += s1[t + off]; s2[t] += s2[t + off]; }
    __syncthreads();
  }
  if (t == 0) {
    float m = s1[0] / (float)Mrows;
    float v = s2[0] / (float)Mrows - m * m;
    mu[j] = m; rstd[j] = rsqrtf(v + EPSV);
  }
}

__global__ void k_bn_apply(float* __restrict__ x, long n,
                           const float* __restrict__ mu, const float* __restrict__ rstd,
                           const float* __restrict__ g, const float* __restrict__ b) {
  long i = (long)blockIdx.x * 256 + threadIdx.x;
  if (i >= n) return;
  int j = (int)(i & 255);
  x[i] = (x[i] - mu[j]) * rstd[j] * g[j] + b[j];
}

// layernorm over H=256, one block per row, bf16 output
__global__ void k_ln(const float* __restrict__ x, const float* __restrict__ g,
                     const float* __restrict__ b, __bf16* __restrict__ outb) {
  int r = blockIdx.x, t = threadIdx.x;
  float v = x[(size_t)r * HD + t];
  __shared__ float s1[256], s2[256];
  s1[t] = v; s2[t] = v * v; __syncthreads();
  for (int off = 128; off > 0; off >>= 1) {
    if (t < off) { s1[t] += s1[t + off]; s2[t] += s2[t + off]; }
    __syncthreads();
  }
  float mu = s1[0] * (1.0f / HD);
  float var = s2[0] * (1.0f / HD) - mu * mu;
  float y = (v - mu) * rsqrtf(var + EPSV) * g[t] + b[t];
  outb[(size_t)r * HD + t] = f2bf(y);
}

// gather root rows: hr_f32[s] = h[s*KSUB], hroot_bf16[s] = bf16(h[s*KSUB])
__global__ void k_gather_roots(const float* __restrict__ h, float* __restrict__ hr,
                               __bf16* __restrict__ hrb) {
  long i = (long)blockIdx.x * 256 + threadIdx.x;
  if (i >= (long)SSUB * HD) return;
  int s = (int)(i >> 8), j = (int)(i & 255);
  float v = h[((size_t)s * KSUB) * HD + j];
  hr[i] = v; hrb[i] = f2bf(v);
}

// m=4 token multi-head attention per node; qkv f32 [S,768]; out bf16 [S,256]
__global__ __launch_bounds__(256)
void k_attn(const float* __restrict__ qkv, __bf16* __restrict__ outb) {
  int n = blockIdx.x, t = threadIdx.x;
  __shared__ float sq[4 * HD], sk[4 * HD], sv[4 * HD];
  __shared__ float sc[64];
  for (int idx = t; idx < 4 * HD; idx += 256) {
    int v = idx >> 8, j = idx & 255;
    size_t base = (size_t)(n * 4 + v) * (3 * HD);
    sq[idx] = qkv[base + j];
    sk[idx] = qkv[base + HD + j];
    sv[idx] = qkv[base + 2 * HD + j];
  }
  __syncthreads();
  if (t < 64) {
    int hh = t >> 4, qj = (t >> 2) & 3, kj = t & 3;
    float s = 0.0f;
#pragma unroll
    for (int d = 0; d < DHEAD; ++d)
      s += sq[qj * HD + hh * DHEAD + d] * sk[kj * HD + hh * DHEAD + d];
    sc[t] = s * 0.125f;  // 1/sqrt(64)
  }
  __syncthreads();
  if (t < 16) {
    int hh = t >> 2, qj = t & 3;
    float mx = -1e30f;
#pragma unroll
    for (int kj = 0; kj < 4; ++kj) mx = fmaxf(mx, sc[hh * 16 + qj * 4 + kj]);
    float sum = 0.0f, e[4];
#pragma unroll
    for (int kj = 0; kj < 4; ++kj) { e[kj] = expf(sc[hh * 16 + qj * 4 + kj] - mx); sum += e[kj]; }
    float inv = 1.0f / sum;
#pragma unroll
    for (int kj = 0; kj < 4; ++kj) sc[hh * 16 + qj * 4 + kj] = e[kj] * inv;
  }
  __syncthreads();
  int hh = t >> 6, dd = t & 63;
#pragma unroll
  for (int qj = 0; qj < 4; ++qj) {
    float o = 0.0f;
#pragma unroll
    for (int kj = 0; kj < 4; ++kj)
      o += sc[hh * 16 + qj * 4 + kj] * sv[kj * HD + hh * DHEAD + dd];
    outb[(size_t)(n * 4 + qj) * HD + hh * DHEAD + dd] = f2bf(o);
  }
}

// mean over m=4 views -> bf16 [NTOT,H]
__global__ void k_mean_m(const float* __restrict__ xa2, __bf16* __restrict__ vvb) {
  long i = (long)blockIdx.x * 256 + threadIdx.x;
  if (i >= (long)NTOT * HD) return;
  int n = (int)(i >> 8), j = (int)(i & 255);
  float s = 0.0f;
#pragma unroll
  for (int v = 0; v < 4; ++v) s += xa2[(size_t)(n * 4 + v) * HD + j];
  vvb[i] = f2bf(s * 0.25f);
}

// out = relu(skip + h1 + h2node[cid] + vv[cid] + kk[f/8]) * valid
__global__ void k_combine(const float* __restrict__ skip, const float* __restrict__ h1,
                          const float* __restrict__ h2n, const float* __restrict__ vv,
                          const float* __restrict__ kk, const int* __restrict__ node_ids,
                          const unsigned char* __restrict__ valid,
                          float* __restrict__ h, __bf16* __restrict__ hb) {
  long i = (long)blockIdx.x * 256 + threadIdx.x;
  if (i >= (long)FN * HD) return;
  int f = (int)(i >> 8), j = (int)(i & 255);
  int cid = node_ids[f]; if (cid < 0) cid = 0;
  float vf = valid[f] ? 1.0f : 0.0f;
  float v = skip[i] + h1[i] + h2n[(size_t)cid * HD + j] + vv[(size_t)cid * HD + j]
          + kk[(size_t)(f >> 3) * HD + j];
  v = fmaxf(v, 0.0f) * vf;
  h[i] = v; hb[i] = f2bf(v);
}

__global__ void k_graphsum(const float* __restrict__ node_embs, const int* __restrict__ batch,
                           float* __restrict__ out) {
  long i = (long)blockIdx.x * 256 + threadIdx.x;
  if (i >= (long)NTOT * HD) return;
  int n = (int)(i >> 8), j = (int)(i & 255);
  atomicAdd(&out[(size_t)batch[n] * HD + j], node_embs[i]);
}

// ---------------------------------------------------------------------------
extern "C" void kernel_launch(void* const* d_in, const int* in_sizes, int n_in,
                              void* d_out, int out_size, void* d_ws, size_t ws_size,
                              hipStream_t stream) {
  // ---- inputs (params tree-flattened: dicts in sorted-key order) ----
  const float* atom_emb = (const float*)d_in[0];
  const float* bond_emb = (const float*)d_in[1];
  const float* fb1   = (const float*)d_in[2];
  const float* fb2   = (const float*)d_in[3];
  const float* fw1   = (const float*)d_in[4];
  const float* fw2   = (const float*)d_in[5];
  const float* gb1   = (const float*)d_in[6];
  const float* gb2   = (const float*)d_in[7];
  const float* gbn_b = (const float*)d_in[8];
  const float* gbn_g = (const float*)d_in[9];
  const float* gw1   = (const float*)d_in[10];
  const float* gw2   = (const float*)d_in[11];
  const float* in_b  = (const float*)d_in[12];
  const float* in_w  = (const float*)d_in[13];
  const float* kk_b  = (const float*)d_in[14];
  const float* kk_w  = (const float*)d_in[15];
  const float* lb1   = (const float*)d_in[16];
  const float* lb2   = (const float*)d_in[17];
  const float* lbn_b = (const float*)d_in[18];
  const float* lbn_g = (const float*)d_in[19];
  const float* ln1_b = (const float*)d_in[20];
  const float* ln1_g = (const float*)d_in[21];
  const float* ln2_b = (const float*)d_in[22];
  const float* ln2_g = (const float*)d_in[23];
  const float* lw1   = (const float*)d_in[24];
  const float* lw2   = (const float*)d_in[25];
  const float* out_b = (const float*)d_in[26];
  const float* out_w = (const float*)d_in[27];
  const float* skip_b= (const float*)d_in[28];
  const float* skip_w= (const float*)d_in[29];
  const float* vv_b  = (const float*)d_in[30];
  const float* vv_w  = (const float*)d_in[31];
  const float* role_emb = (const float*)d_in[32];
  const int* xid        = (const int*)d_in[33];
  const int* ea_global  = (const int*)d_in[34];
  const int* ea_intra   = (const int*)d_in[35];
  const int* intra_ei   = (const int*)d_in[36];
  const int* edge_index = (const int*)d_in[37];
  const int* node_ids   = (const int*)d_in[38];
  const unsigned char* valid = (const unsigned char*)d_in[39];
  const int* batch      = (const int*)d_in[41];
  float* out = (float*)d_out;

  const int* src_l = intra_ei;
  const int* dst_l = intra_ei + EINTRA;
  const int* src_g = edge_index;
  const int* dst_g = edge_index + EGLOB;

  // ---- workspace bump allocator ----
  char* ws = (char*)d_ws;
  size_t off = 0;
  auto alloc = [&](size_t bytes) -> void* {
    void* p = ws + off;
    off = (off + bytes + 255) & ~(size_t)255;
    return p;
  };
  const size_t HH = (size_t)HD * HD;

  // bf16 weight pool (all layers)
  __bf16* w_lw1  = (__bf16*)alloc(LLAYERS * HH * 2);
  __bf16* w_lw2  = (__bf16*)alloc(LLAYERS * HH * 2);
  __bf16* w_gw1  = (__bf16*)alloc(LLAYERS * HH * 2);
  __bf16* w_gw2  = (__bf16*)alloc(LLAYERS * HH * 2);
  __bf16* w_inw  = (__bf16*)alloc(LLAYERS * HD * 3 * HD * 2);
  __bf16* w_outw = (__bf16*)alloc(LLAYERS * HH * 2);
  __bf16* w_fw1  = (__bf16*)alloc(LLAYERS * HD * 4 * HD * 2);
  __bf16* w_fw2  = (__bf16*)alloc(LLAYERS * 4 * HD * HD * 2);
  __bf16* w_vvw  = (__bf16*)alloc(LLAYERS * HH * 2);
  __bf16* w_skw  = (__bf16*)alloc(LLAYERS * HH * 2);
  __bf16* w_kkw  = (__bf16*)alloc(LLAYERS * HH * 2);

  float*  h      = (float*)alloc((size_t)FN * HD * 4);
  __bf16* hb     = (__bf16*)alloc((size_t)FN * HD * 2);
  float*  skip   = (float*)alloc((size_t)FN * HD * 4);
  float*  agg    = (float*)alloc((size_t)FN * HD * 4);
  __bf16* aggb   = (__bf16*)alloc((size_t)FN * HD * 2);
  __bf16* t1b    = (__bf16*)alloc((size_t)FN * HD * 2);
  float*  h1     = (float*)alloc((size_t)FN * HD * 4);
  float*  xsum   = (float*)alloc((size_t)NTOT * HD * 4);
  float*  cnt    = (float*)alloc((size_t)NTOT * 4);
  __bf16* xsumb  = (__bf16*)alloc((size_t)NTOT * HD * 2);
  float*  aggg   = (float*)alloc((size_t)NTOT * HD * 4);
  __bf16* agggb  = (__bf16*)alloc((size_t)NTOT * HD * 2);
  __bf16* g1b    = (__bf16*)alloc((size_t)NTOT * HD * 2);
  float*  hg     = (float*)alloc((size_t)NTOT * HD * 4);
  float*  hr     = (float*)alloc((size_t)SSUB * HD * 4);
  __bf16* hrootb = (__bf16*)alloc((size_t)SSUB * HD * 2);
  __bf16* rlnb   = (__bf16*)alloc((size_t)SSUB * HD * 2);
  float*  qkv    = (float*)alloc((size_t)SSUB * 3 * HD * 4);
  __bf16* ob     = (__bf16*)alloc((size_t)SSUB * HD * 2);
  float*  xa     = (float*)alloc((size_t)SSUB * HD * 4);
  __bf16* ln2b   = (__bf16*)alloc((size_t)SSUB * HD * 2);
  __bf16* f1b    = (__bf16*)alloc((size_t)SSUB * 4 * HD * 2);
  float*  xa2    = (float*)alloc((size_t)SSUB * HD * 4);
  __bf16* vvnb   = (__bf16*)alloc((size_t)NTOT * HD * 2);
  float*  vvout  = (float*)alloc((size_t)NTOT * HD * 4);
  float*  kkout  = (float*)alloc((size_t)SSUB * HD * 4);
  float*  bn_mu  = (float*)alloc(HD * 4);
  float*  bn_rs  = (float*)alloc(HD * 4);

  auto blocks = [](long n) { return (unsigned)((n + 255) / 256); };
  auto cvt = [&](const float* s, __bf16* d, long n) {
    k_cvt_bf16<<<blocks(n), 256, 0, stream>>>(s, d, n);
  };
  auto gemm = [&](const __bf16* A, const __bf16* W, const float* bias,
                  const float* resid, float* Cf, __bf16* Cb,
                  int Mm, int Nn, int Kk, int act) {
    dim3 g(Nn / 64, Mm / 128);
    k_gemm_wmma<<<g, 256, 0, stream>>>(A, W, bias, resid, Cf, Cb, Mm, Nn, Kk, act);
  };

  // ---- weight conversion to bf16 (once per call) ----
  cvt(lw1,  w_lw1,  LLAYERS * (long)HH);
  cvt(lw2,  w_lw2,  LLAYERS * (long)HH);
  cvt(gw1,  w_gw1,  LLAYERS * (long)HH);
  cvt(gw2,  w_gw2,  LLAYERS * (long)HH);
  cvt(in_w, w_inw,  LLAYERS * (long)HD * 3 * HD);
  cvt(out_w,w_outw, LLAYERS * (long)HH);
  cvt(fw1,  w_fw1,  LLAYERS * (long)HD * 4 * HD);
  cvt(fw2,  w_fw2,  LLAYERS * (long)4 * HD * HD);
  cvt(vv_w, w_vvw,  LLAYERS * (long)HH);
  cvt(skip_w,w_skw, LLAYERS * (long)HH);
  cvt(kk_w, w_kkw,  LLAYERS * (long)HH);

  // ---- constants: valid-counts per original node ----
  k_zero<<<blocks(NTOT), 256, 0, stream>>>(cnt, NTOT);
  k_counts<<<blocks(FN), 256, 0, stream>>>(node_ids, valid, cnt);

  // ---- initial embedding ----
  k_init_h<<<blocks((long)FN * HD), 256, 0, stream>>>(atom_emb, role_emb, xid,
                                                      node_ids, valid, h, hb);

  for (int l = 0; l < LLAYERS; ++l) {
    const size_t oH  = (size_t)l * HD;
    const size_t oHH = (size_t)l * HH;
    const size_t o3  = (size_t)l * 3 * HD;
    const size_t o4  = (size_t)l * 4 * HD;

    // skip projection
    gemm(hb, w_skw + oHH, skip_b + oH, nullptr, skip, nullptr, FN, HD, HD, 0);

    // local GINE: agg = h + segsum(relu(h[src]+ea))   (agg=h copy via TDM)
    k_copy_tdm<<<FN / 32, 32, 0, stream>>>(h, agg);
    k_edge_msg<<<blocks((long)EINTRA * HD), 256, 0, stream>>>(h, bond_emb, src_l,
                                                              dst_l, ea_intra, agg, EINTRA);
    cvt(agg, aggb, (long)FN * HD);
    gemm(aggb, w_lw1 + oHH, lb1 + oH, nullptr, nullptr, t1b, FN, HD, HD, 1);
    gemm(t1b, w_lw2 + oHH, lb2 + oH, nullptr, h1, nullptr, FN, HD, HD, 0);
    k_bn_stats<<<HD, 256, 0, stream>>>(h1, FN, bn_mu, bn_rs);
    k_bn_apply<<<blocks((long)FN * HD), 256, 0, stream>>>(h1, (long)FN * HD, bn_mu,
                                                          bn_rs, lbn_g + oH, lbn_b + oH);

    // global GINE on original nodes
    k_zero<<<blocks((long)NTOT * HD), 256, 0, stream>>>(xsum, (long)NTOT * HD);
    k_segsum<<<blocks((long)FN * HD), 256, 0, stream>>>(h, node_ids, xsum);
    k_divcnt<<<blocks((long)NTOT * HD), 256, 0, stream>>>(xsum, cnt, xsumb);
    k_copy_tdm<<<NTOT / 32, 32, 0, stream>>>(xsum, aggg);
    k_edge_msg<<<blocks((long)EGLOB * HD), 256, 0, stream>>>(xsum, bond_emb, src_g,
                                                             dst_g, ea_global, aggg, EGLOB);
    cvt(aggg, agggb, (long)NTOT * HD);
    gemm(agggb, w_gw1 + oHH, gb1 + oH, nullptr, nullptr, g1b, NTOT, HD, HD, 1);
    gemm(g1b, w_gw2 + oHH, gb2 + oH, nullptr, hg, nullptr, NTOT, HD, HD, 0);
    k_bn_stats<<<HD, 256, 0, stream>>>(hg, NTOT, bn_mu, bn_rs);
    k_bn_apply<<<blocks((long)NTOT * HD), 256, 0, stream>>>(hg, (long)NTOT * HD, bn_mu,
                                                            bn_rs, gbn_g + oH, gbn_b + oH);

    // view attention over m=4 root tokens
    k_gather_roots<<<blocks((long)SSUB * HD), 256, 0, stream>>>(h, hr, hrootb);
    k_ln<<<SSUB, 256, 0, stream>>>(hr, ln1_g + oH, ln1_b + oH, rlnb);
    gemm(rlnb, w_inw + (size_t)l * HD * 3 * HD, in_b + o3, nullptr, qkv, nullptr,
         SSUB, 3 * HD, HD, 0);
    k_attn<<<NTOT, 256, 0, stream>>>(qkv, ob);
    gemm(ob, w_outw + oHH, out_b + oH, hr, xa, nullptr, SSUB, HD, HD, 0);
    k_ln<<<SSUB, 256, 0, stream>>>(xa, ln2_g + oH, ln2_b + oH, ln2b);
    gemm(ln2b, w_fw1 + (size_t)l * HD * 4 * HD, fb1 + o4, nullptr, nullptr, f1b,
         SSUB, 4 * HD, HD, 2);
    gemm(f1b, w_fw2 + (size_t)l * 4 * HD * HD, fb2 + oH, xa, xa2, nullptr,
         SSUB, HD, 4 * HD, 0);
    k_mean_m<<<blocks((long)NTOT * HD), 256, 0, stream>>>(xa2, vvnb);
    gemm(vvnb, w_vvw + oHH, vv_b + oH, nullptr, vvout, nullptr, NTOT, HD, HD, 0);
    gemm(hrootb, w_kkw + oHH, kk_b + oH, nullptr, kkout, nullptr, SSUB, HD, HD, 0);

    // combine + relu + valid mask
    k_combine<<<blocks((long)FN * HD), 256, 0, stream>>>(skip, h1, hg, vvout, kkout,
                                                         node_ids, valid, h, hb);
  }

  // ---- readout: seg_mean then graph segment_sum ----
  k_zero<<<blocks((long)NTOT * HD), 256, 0, stream>>>(xsum, (long)NTOT * HD);
  k_segsum<<<blocks((long)FN * HD), 256, 0, stream>>>(h, node_ids, xsum);
  k_divcnt<<<blocks((long)NTOT * HD), 256, 0, stream>>>(xsum, cnt, nullptr);
  k_zero<<<blocks((long)NGRAPH * HD), 256, 0, stream>>>(out, (long)NGRAPH * HD);
  k_graphsum<<<blocks((long)NTOT * HD), 256, 0, stream>>>(xsum, batch, out);
}
